// PTB2bBmm_82987358094192
// MI455X (gfx1250) — matrix operand order repaired and checked by reading, not compiled, not run on hardware
//
#include <hip/hip_runtime.h>

// ---------------------------------------------------------------------------
// score = sigmoid(alpha0 * q @ k^T + bias); out = score @ v
// B=16, M=N=2048, K=D=128.  f32 in/out; internally f16 WMMA (16x16x32).
// Bias stream (268MB @ 23.3TB/s ~= 13.7us) is the roofline floor; k/vT
// (f16, 8.4MB each) stay L2-resident. Per-lane base pointers + imm offsets;
// sched_barrier-pinned software pipeline keeps 8-16 loads in flight under
// every WMMA chain; sigmoid uses hardware v_exp_f32 + v_rcp_f32.
// ---------------------------------------------------------------------------

typedef _Float16 v16h __attribute__((ext_vector_type(16)));
typedef float    v8f  __attribute__((ext_vector_type(8)));

#if __has_builtin(__builtin_amdgcn_sched_barrier)
#define SCHED_FENCE() __builtin_amdgcn_sched_barrier(0)
#else
#define SCHED_FENCE()
#endif

#if __has_builtin(__builtin_amdgcn_rcpf)
#define FAST_RCP(x) __builtin_amdgcn_rcpf(x)
#else
#define FAST_RCP(x) (1.0f / (x))
#endif

#define B_ 16
#define M_ 2048
#define N_ 2048
#define K_ 128
#define D_ 128
#define ALPHA0 0.08838834764831845f /* 1/sqrt(128) */

#define MTILE  128   // per block: 8 waves x 16 rows
#define NTILE  64
#define NWAVES 8

union AFrag { v16h v; uint4 u[2]; };

// ---- pre-pass: f32 -> f16 with scale (q gets alpha0 folded in) ----
__global__ void cvt_scale_kernel(const float* __restrict__ src,
                                 _Float16* __restrict__ dst,
                                 int n, float scale) {
    int i = blockIdx.x * blockDim.x + threadIdx.x;
    if (i < n) dst[i] = (_Float16)(src[i] * scale);
}

// ---- pre-pass: v (B,N,D) f32 -> vt (B,D,N) f16 (writes coalesced over n) ----
__global__ void cvt_transpose_v_kernel(const float* __restrict__ v,
                                       _Float16* __restrict__ vt) {
    size_t i = (size_t)blockIdx.x * blockDim.x + threadIdx.x;
    const size_t total = (size_t)B_ * D_ * N_;
    if (i >= total) return;
    int n = (int)(i % N_);
    size_t t = i / N_;
    int d = (int)(t % D_);
    int b = (int)(t / D_);
    vt[i] = (_Float16)v[((size_t)b * N_ + n) * D_ + d];
}

// ---- fused sigmoid-attention kernel ----
__global__ __launch_bounds__(256) void sattn_wmma_kernel(
    const _Float16* __restrict__ qs,   // (B,M,K)  pre-scaled by alpha0
    const _Float16* __restrict__ kb,   // (B,N,K)
    const _Float16* __restrict__ vt,   // (B,D,N)  transposed V
    const float*    __restrict__ bias, // (B,M,N)
    float* __restrict__ out)           // (B,M,D)
{
    __shared__ alignas(16) _Float16 Pbuf[NWAVES][16][NTILE]; // 16 KB

    const int tid  = threadIdx.x;
    const int w    = tid >> 5;        // wave id 0..7
    const int lane = tid & 31;
    const int half = lane >> 4;       // 0/1 (lane half per WMMA layouts)
    const int l16  = lane & 15;

    const int mblk = blockIdx.x;
    const int b    = blockIdx.y;
    const int m0   = mblk * MTILE + w * 16;   // this wave's 16 M-rows

    // ---- Q A-fragments for K=128: 4 fragments of 16x32 f16, kept in VGPRs ----
    AFrag aq[4];
    {
        const _Float16* qp = qs + ((size_t)b * M_ + (m0 + l16)) * K_ + 16 * half;
        #pragma unroll
        for (int kk = 0; kk < 4; ++kk) {
            aq[kk].u[0] = *(const uint4*)(qp + 32 * kk);       // K koff..koff+7
            aq[kk].u[1] = *(const uint4*)(qp + 32 * kk + 8);   // K koff+8..+15
        }
    }

    // ---- O accumulators: 8 fragments (16x16 f32) covering D=128 ----
    v8f oacc[8];
    #pragma unroll
    for (int g = 0; g < 8; ++g) oacc[g] = (v8f){};

    // ---- per-lane streaming base pointers (advance once per N tile) ----
    const float* bptr = bias + ((size_t)b * M_ + m0 + 8 * half) * N_ + l16;
    const _Float16* kptr = kb + ((size_t)b * N_ + l16) * K_ + 16 * half;
    const _Float16* vptr = vt + ((size_t)b * D_ + l16) * N_ + 16 * half;

    // ---- per-lane LDS pointers (immediate DS offsets inside the loop) ----
    _Float16*       pwr = &Pbuf[w][8 * half][l16]; // + r*NTILE + 16*j
    const _Float16* prd = &Pbuf[w][l16][16 * half]; // + 32*kk2 (+8 hi chunk)

    for (int nt = 0; nt < N_; nt += NTILE) {
        // ---- bias fragments for all 4 column groups up front (HBM stream) ----
        v8f accs[4];
        #pragma unroll
        for (int j = 0; j < 4; ++j)
            #pragma unroll
            for (int r = 0; r < 8; ++r)
                accs[j][r] = bptr[(size_t)r * N_ + 16 * j];   // imm offsets

        // ---- GEMM1: S = Qs @ K^T + bias ; P = sigmoid(S) -> LDS strip ----
        // 2-deep double buffer on the B fragments, sched_barrier-pinned so
        // group j+1's 8 b128 loads are in flight under group j's WMMA chain.
        v16h bf[2][4];
        #pragma unroll
        for (int kk = 0; kk < 4; ++kk)
            bf[0][kk] = *(const v16h*)(kptr + 32 * kk);

        #pragma unroll
        for (int j = 0; j < 4; ++j) {
            const int cur = j & 1, nxt = cur ^ 1;
            if (j < 3) {
                #pragma unroll
                for (int kk = 0; kk < 4; ++kk)
                    bf[nxt][kk] = *(const v16h*)(kptr + 16 * (j + 1) * K_
                                                      + 32 * kk);
            }
            SCHED_FENCE();  // loads above stay above the WMMA chain below

            #pragma unroll
            for (int kk = 0; kk < 4; ++kk)
                accs[j] = __builtin_amdgcn_wmma_f32_16x16x32_f16(
                    false, aq[kk].v, false, bf[cur][kk], (short)0, accs[j],
                    false, false);

            // sigmoid via v_exp_f32 + v_rcp_f32 (no IEEE div expansion),
            // f16 convert, store into this wave's LDS P strip
            #pragma unroll
            for (int r = 0; r < 8; ++r) {
                float p = FAST_RCP(1.0f + __expf(-accs[j][r]));
                pwr[r * NTILE + 16 * j] = (_Float16)p;        // ds imm offsets
            }
        }
        // per-wave LDS strip: DS is in-order within a wave; fence stops the
        // compiler from hoisting the reads above the stores (no block barrier)
        __builtin_amdgcn_fence(__ATOMIC_ACQ_REL, "wavefront");

        // ---- GEMM2: O += P @ V  (contraction over this 64-wide N tile) ----
        #pragma unroll
        for (int kk2 = 0; kk2 < NTILE / 32; ++kk2) {
            AFrag ap;
            ap.u[0] = *(const uint4*)(prd + 32 * kk2);        // ds_load_b128
            ap.u[1] = *(const uint4*)(prd + 32 * kk2 + 8);    // ds_load_b128

            // preload all 8 B fragments for this chunk: 16 b128 in flight
            v16h bv[8];
            #pragma unroll
            for (int g = 0; g < 8; ++g)
                bv[g] = *(const v16h*)(vptr + (size_t)(16 * g) * N_
                                            + 32 * kk2);      // imm offsets
            SCHED_FENCE();  // keep the 16-load clause above the WMMA chain

            #pragma unroll
            for (int g = 0; g < 8; ++g)
                oacc[g] = __builtin_amdgcn_wmma_f32_16x16x32_f16(
                    false, ap.v, false, bv[g], (short)0, oacc[g],
                    false, false);
        }
        __builtin_amdgcn_fence(__ATOMIC_ACQ_REL, "wavefront");

        bptr += NTILE;          // next bias tile
        kptr += NTILE * K_;     // next 64 k-rows
        vptr += NTILE;          // next 64 columns of vT
    }

    // ---- epilogue: C-layout fragments -> (B,M,D) f32, coalesced per half ----
    float* optr = out + ((size_t)b * M_ + m0 + 8 * half) * D_ + l16;
    #pragma unroll
    for (int g = 0; g < 8; ++g)
        #pragma unroll
        for (int r = 0; r < 8; ++r)
            optr[(size_t)r * D_ + 16 * g] = oacc[g][r];
}

extern "C" void kernel_launch(void* const* d_in, const int* in_sizes, int n_in,
                              void* d_out, int out_size, void* d_ws, size_t ws_size,
                              hipStream_t stream) {
    (void)in_sizes; (void)n_in; (void)out_size;

    const float* q    = (const float*)d_in[0];
    const float* k    = (const float*)d_in[1];
    const float* v    = (const float*)d_in[2];
    const float* bias = (const float*)d_in[3];
    float* out = (float*)d_out;

    const size_t SZ = (size_t)B_ * M_ * K_;   // == B*N*K == B*D*N = 4,194,304
    if (ws_size < 3 * SZ * sizeof(_Float16)) return;  // need ~25 MB scratch

    _Float16* qs = (_Float16*)d_ws;
    _Float16* kb = qs + SZ;
    _Float16* vt = kb + SZ;

    const int threads = 256;
    const int blocks  = (int)(SZ / threads);
    cvt_scale_kernel<<<blocks, threads, 0, stream>>>(q, qs, (int)SZ, ALPHA0);
    cvt_scale_kernel<<<blocks, threads, 0, stream>>>(k, kb, (int)SZ, 1.0f);
    cvt_transpose_v_kernel<<<blocks, threads, 0, stream>>>(v, vt);

    dim3 grid(M_ / MTILE, B_);
    sattn_wmma_kernel<<<grid, 256, 0, stream>>>(qs, kb, vt, bias, out);
}
